// GraphSAGERegressor_4243427688731
// MI455X (gfx1250) — matrix operand order repaired and checked by reading, not compiled, not run on hardware
//
#include <hip/hip_runtime.h>
#include <hip/hip_bf16.h>

typedef __attribute__((ext_vector_type(16))) _Float16 v16h;
typedef __attribute__((ext_vector_type(8)))  float    v8f;

// ---------------------------------------------------------------- zero
__global__ void zero_kernel(float* __restrict__ p, long n) {
  long i = (long)blockIdx.x * blockDim.x + threadIdx.x;
  long stride = (long)gridDim.x * blockDim.x;
  for (; i < n; i += stride) p[i] = 0.0f;
}

// ---------------------------------------------------------------- degree count
__global__ void deg_kernel(const int* __restrict__ dst, float* __restrict__ cnt, int E) {
  int e = blockIdx.x * blockDim.x + threadIdx.x;
  if (e < E)
    __hip_atomic_fetch_add(&cnt[dst[e]], 1.0f, __ATOMIC_RELAXED, __HIP_MEMORY_SCOPE_AGENT);
}

// ---------------------------------------------------------------- per-graph node count
__global__ void poolcnt_kernel(const int* __restrict__ batch, float* __restrict__ pc, int N) {
  int n = blockIdx.x * blockDim.x + threadIdx.x;
  if (n < N)
    __hip_atomic_fetch_add(&pc[batch[n]], 1.0f, __ATOMIC_RELAXED, __HIP_MEMORY_SCOPE_AGENT);
}

// ---------------------------------------------------------------- edge scatter-add
// one wave (32 lanes) per edge; each lane moves 2 of the 64 channels
__global__ void scatter_kernel(const float* __restrict__ vals,
                               const int* __restrict__ src,
                               const int* __restrict__ dst,
                               float* __restrict__ accum, int E) {
  int t = blockIdx.x * blockDim.x + threadIdx.x;
  int e = t >> 5;
  int lane = t & 31;
  if (e < E) {
    int s = src[e];
    int d = dst[e];
    float2 v = *(const float2*)(vals + (long)s * 64 + lane * 2);
    float* o = accum + (long)d * 64 + lane * 2;
    __hip_atomic_fetch_add(o + 0, v.x, __ATOMIC_RELAXED, __HIP_MEMORY_SCOPE_AGENT);
    __hip_atomic_fetch_add(o + 1, v.y, __ATOMIC_RELAXED, __HIP_MEMORY_SCOPE_AGENT);
  }
}

// ---------------------------------------------------------------- fused SAGE layer
// h = relu( (aggsum/deg) @ Wl + xin @ Wr + bias )
// computed as [aggmean | xin] (16x128, f16) @ [Wl ; Wr] (128x64, f16) per 16-node tile,
// one wave32 per tile, 16x v_wmma_f32_16x16x32_f16 per tile, f32 accumulate.
// POOL=false: write h to hout.  POOL=true: relu'd rows atomically mean-pooled by graph.
template<bool POOL>
__global__ __launch_bounds__(512)
void sage_gemm(const float* __restrict__ aggsum,
               const float* __restrict__ cnt,
               const float* __restrict__ xin,
               const float* __restrict__ Wl,
               const float* __restrict__ Wr,
               const float* __restrict__ bias,
               float* __restrict__ hout,
               float* __restrict__ pool,
               const int* __restrict__ batch,
               int ntiles, int N) {
  // B operand pre-swizzled into the exact 16-bit B (32x16) WMMA lane layout:
  // index = ((kt*4 + nt)*32 + lane)*16 + j
  // element (K = kt*32 + (lane>=16 ? 8 : 0) + j + (j&8) , N = nt*16 + (lane&15))
  __shared__ __attribute__((aligned(32))) _Float16 Blds[8192];
  for (int i = threadIdx.x; i < 8192; i += blockDim.x) {
    int kt   = i >> 11;
    int nt   = (i >> 9) & 3;
    int lane = (i >> 4) & 31;
    int j    = i & 15;
    int K = kt * 32 + ((lane >= 16) ? 8 : 0) + j + (j & 8);
    int n = nt * 16 + (lane & 15);
    float w = (K < 64) ? Wl[K * 64 + n] : Wr[(K - 64) * 64 + n];
    Blds[i] = (_Float16)w;
  }
  __syncthreads();

  int wave = threadIdx.x >> 5;
  int lane = threadIdx.x & 31;
  int tile = blockIdx.x * (blockDim.x >> 5) + wave;
  if (tile >= ntiles) return;   // wave-uniform: EXEC stays all-1s for WMMA

  int row      = lane & 15;
  int half_sel = lane >> 4;      // lanes 16..31 hold the upper K halves
  int koff     = half_sel * 8;
  int node     = tile * 16 + row;
  if (node >= N) node = N - 1;   // clamp loads only (partial-tile safety)

  float inv = 1.0f / fmaxf(cnt[node], 1.0f);

  // A operand: 16-bit A 16x32 layout; 4 K-tiles (K=0..63 aggmean, K=64..127 xin)
  v16h a[4];
  const float* abase = aggsum + (long)node * 64;
  const float* xbase = xin    + (long)node * 64;
#pragma unroll
  for (int kt = 0; kt < 4; ++kt) {
    const float* s = (kt < 2) ? (abase + kt * 32) : (xbase + (kt - 2) * 32);
    float sc = (kt < 2) ? inv : 1.0f;
    float4 f0 = *(const float4*)(s + koff);
    float4 f1 = *(const float4*)(s + koff + 4);
    float4 f2 = *(const float4*)(s + 16 + koff);
    float4 f3 = *(const float4*)(s + 16 + koff + 4);
    a[kt][0]  = (_Float16)(f0.x * sc); a[kt][1]  = (_Float16)(f0.y * sc);
    a[kt][2]  = (_Float16)(f0.z * sc); a[kt][3]  = (_Float16)(f0.w * sc);
    a[kt][4]  = (_Float16)(f1.x * sc); a[kt][5]  = (_Float16)(f1.y * sc);
    a[kt][6]  = (_Float16)(f1.z * sc); a[kt][7]  = (_Float16)(f1.w * sc);
    a[kt][8]  = (_Float16)(f2.x * sc); a[kt][9]  = (_Float16)(f2.y * sc);
    a[kt][10] = (_Float16)(f2.z * sc); a[kt][11] = (_Float16)(f2.w * sc);
    a[kt][12] = (_Float16)(f3.x * sc); a[kt][13] = (_Float16)(f3.y * sc);
    a[kt][14] = (_Float16)(f3.z * sc); a[kt][15] = (_Float16)(f3.w * sc);
  }

  // 16 WMMAs as one dense block (B streamed from LDS as ds_load_b128 pairs)
  v8f acc[4];
#pragma unroll
  for (int nt = 0; nt < 4; ++nt) {
    v8f c = {};
#pragma unroll
    for (int kt = 0; kt < 4; ++kt) {
      const v16h* bp = (const v16h*)&Blds[((kt * 4 + nt) * 32 + lane) * 16];
      c = __builtin_amdgcn_wmma_f32_16x16x32_f16(
          /*neg_a=*/false, a[kt], /*neg_b=*/false, *bp,
          /*c_mod=*/(short)0, c, /*reuse_a=*/false, /*reuse_b=*/false);
    }
    acc[nt] = c;
  }

  // epilogue: D layout -> lane = col N (within n-tile), VGPR r = row rowbase+r
  int col     = lane & 15;
  int rowbase = half_sel * 8;
  bool full   = (tile * 16 + 16 <= N);   // wave-uniform: scalar branch only

  if (full) {
#pragma unroll
    for (int nt = 0; nt < 4; ++nt) {
      int c = nt * 16 + col;
      float bv = bias[c];
#pragma unroll
      for (int r = 0; r < 8; ++r) {
        int m = tile * 16 + rowbase + r;
        float v = fmaxf(acc[nt][r] + bv, 0.0f);
        if (!POOL) {
          hout[(long)m * 64 + c] = v;
        } else {
          int g = batch[m];
          __hip_atomic_fetch_add(&pool[(long)g * 64 + c], v,
                                 __ATOMIC_RELAXED, __HIP_MEMORY_SCOPE_AGENT);
        }
      }
    }
  } else {
#pragma unroll
    for (int nt = 0; nt < 4; ++nt) {
      int c = nt * 16 + col;
      float bv = bias[c];
#pragma unroll
      for (int r = 0; r < 8; ++r) {
        int m = tile * 16 + rowbase + r;
        if (m < N) {
          float v = fmaxf(acc[nt][r] + bv, 0.0f);
          if (!POOL) {
            hout[(long)m * 64 + c] = v;
          } else {
            int g = batch[m];
            __hip_atomic_fetch_add(&pool[(long)g * 64 + c], v,
                                   __ATOMIC_RELAXED, __HIP_MEMORY_SCOPE_AGENT);
          }
        }
      }
    }
  }
}

// ---------------------------------------------------------------- readout
// out[g] = (pool_sum[g]/max(cnt,1)) . Wro + bro   — one wave per graph
__global__ void readout_kernel(const float* __restrict__ pool,
                               const float* __restrict__ pc,
                               const float* __restrict__ Wro,
                               const float* __restrict__ bro,
                               float* __restrict__ out, int G) {
  int t = blockIdx.x * blockDim.x + threadIdx.x;
  int g = t >> 5;
  int lane = t & 31;
  if (g < G) {
    const float* p = pool + (long)g * 64;
    float s = p[lane] * Wro[lane] + p[lane + 32] * Wro[lane + 32];
#pragma unroll
    for (int off = 16; off > 0; off >>= 1) s += __shfl_xor(s, off, 32);
    if (lane == 0) out[g] = s / fmaxf(pc[g], 1.0f) + bro[0];
  }
}

// ---------------------------------------------------------------- launcher
extern "C" void kernel_launch(void* const* d_in, const int* in_sizes, int n_in,
                              void* d_out, int out_size, void* d_ws, size_t ws_size,
                              hipStream_t stream) {
  const float* x    = (const float*)d_in[0];
  const float* Wl1  = (const float*)d_in[1];
  const float* bl1  = (const float*)d_in[2];
  const float* Wr1  = (const float*)d_in[3];
  const float* Wl2  = (const float*)d_in[4];
  const float* bl2  = (const float*)d_in[5];
  const float* Wr2  = (const float*)d_in[6];
  const float* Wro  = (const float*)d_in[7];
  const float* bro  = (const float*)d_in[8];
  const int*   eidx = (const int*)d_in[9];
  const int*   batch= (const int*)d_in[10];

  const int N = in_sizes[0] / 64;
  const int E = in_sizes[9] / 2;
  const int G = out_size;
  const int* src = eidx;
  const int* dst = eidx + E;

  float* ws      = (float*)d_ws;
  float* aggsum  = ws;                       // N*64
  float* h1      = aggsum + (long)N * 64;    // N*64
  float* cnt     = h1 + (long)N * 64;        // N
  float* pool    = cnt + N;                  // G*64
  float* poolcnt = pool + (long)G * 64;      // G
  float* out     = (float*)d_out;

  const int ntiles = (N + 15) / 16;
  const int gemm_blocks = (ntiles + 15) / 16;        // 16 waves (512 thr) per block

  // init
  zero_kernel<<<2048, 256, 0, stream>>>(aggsum, (long)N * 64);
  zero_kernel<<<256, 256, 0, stream>>>(cnt, N);
  zero_kernel<<<64, 256, 0, stream>>>(pool, (long)G * 64 + G);  // pool + poolcnt
  // degrees / graph sizes
  deg_kernel<<<(E + 255) / 256, 256, 0, stream>>>(dst, cnt, E);
  poolcnt_kernel<<<(N + 255) / 256, 256, 0, stream>>>(batch, poolcnt, N);
  // layer 1
  scatter_kernel<<<(E + 7) / 8, 256, 0, stream>>>(x, src, dst, aggsum, E);
  sage_gemm<false><<<gemm_blocks, 512, 0, stream>>>(aggsum, cnt, x, Wl1, Wr1, bl1,
                                                    h1, nullptr, nullptr, ntiles, N);
  // layer 2 (+ fused global mean pool)
  zero_kernel<<<2048, 256, 0, stream>>>(aggsum, (long)N * 64);
  scatter_kernel<<<(E + 7) / 8, 256, 0, stream>>>(h1, src, dst, aggsum, E);
  sage_gemm<true><<<gemm_blocks, 512, 0, stream>>>(aggsum, cnt, h1, Wl2, Wr2, bl2,
                                                   nullptr, pool, batch, ntiles, N);
  // readout
  readout_kernel<<<(G * 32 + 255) / 256, 256, 0, stream>>>(pool, poolcnt, Wro, bro, out, G);
}
